// StackLSTM_67800353734750
// MI455X (gfx1250) — matrix operand order repaired and checked by reading, not compiled, not run on hardware
//
#include <hip/hip_runtime.h>
#include <hip/hip_bf16.h>

// ---------------------------------------------------------------------------
// Types for CDNA5 WMMA (gfx1250, wave32)
// ---------------------------------------------------------------------------
typedef __attribute__((ext_vector_type(16))) __bf16 v16bf;
typedef __attribute__((ext_vector_type(8)))  float  v8f;

// Problem constants (match reference)
#define BB   64
#define TT   128
#define EMB  300
#define EP   320          // EMB padded to multiple of 32 (WMMA K step)
#define RNN  256
#define G4   1024
#define OUT  256
#define G3   768
#define H2   512

__device__ __forceinline__ unsigned short f2bf(float f) {
    unsigned int u = __float_as_uint(f);
    unsigned int r = u + 0x7fffu + ((u >> 16) & 1u);   // round-to-nearest-even
    return (unsigned short)(r >> 16);
}
__device__ __forceinline__ float sigm(float x) { return 1.f / (1.f + expf(-x)); }

__device__ __forceinline__ v8f wmma_bf16(v16bf a, v16bf b, v8f c) {
    return __builtin_amdgcn_wmma_f32_16x16x32_bf16(
        /*neg_a=*/false, a, /*neg_b=*/false, b,
        /*c_mod=*/(short)0, c, /*reuse_a=*/false, /*reuse_b=*/false);
}

// A fragment (16x32 bf16, contiguous k order): lane L -> row m = L%16,
// halfs j = K[kb + j], kb = (L<16?0:16)
__device__ __forceinline__ v16bf load_a_frag(const unsigned short* base, int stride, int lane) {
    int m  = lane & 15;
    int kb = (lane < 16) ? 0 : 16;
    const unsigned short* p = base + m * stride + kb;
    union { uint4 q[2]; v16bf v; } u;
    u.q[0] = *(const uint4*)(p);
    u.q[1] = *(const uint4*)(p + 8);
    return u.v;
}

// B fragment (32x16 bf16): lane L -> col n = L%16, halfs j = K[kbase + j],
// kbase = (L<16 ? 0 : 16). Pre-packed contiguously: 32 lanes * 16 halfs / tile.
__device__ __forceinline__ v16bf load_b_frag(const unsigned short* packed, int lane) {
    const unsigned short* p = packed + lane * 16;
    union { uint4 q[2]; v16bf v; } u;
    u.q[0] = *(const uint4*)(p);
    u.q[1] = *(const uint4*)(p + 8);
    return u.v;
}

// ---------------------------------------------------------------------------
// Weight packing: w [Nout x K] f32 row-major -> B-fragment packed bf16
// layout: [kt][nt][lane(32)][j(16)]
// ---------------------------------------------------------------------------
__global__ void k_pack_b(const float* __restrict__ w, unsigned short* __restrict__ out,
                         int Nout, int K, int KT, int NT) {
    int idx = blockIdx.x * blockDim.x + threadIdx.x;
    int total = KT * NT * 512;
    if (idx >= total) return;
    int j    = idx & 15;
    int lane = (idx >> 4) & 31;
    int rest = idx >> 9;
    int nt   = rest % NT;
    int kt   = rest / NT;
    int n = nt * 16 + (lane & 15);
    int k = kt * 32 + ((lane < 16) ? 0 : 16) + j;
    float v = (k < K && n < Nout) ? w[n * K + k] : 0.f;
    out[idx] = f2bf(v);
}

// ---------------------------------------------------------------------------
// Embedding gather -> time-major bf16 X for both directions.
// row = t*64 + b.  Xb uses per-row reversal t' = len-1-t for t < len.
// ---------------------------------------------------------------------------
__global__ void k_gather_x(const int* __restrict__ ids, const int* __restrict__ lengths,
                           const float* __restrict__ emb,
                           unsigned short* __restrict__ Xf, unsigned short* __restrict__ Xb) {
    int idx = blockIdx.x * blockDim.x + threadIdx.x;  // over 8192*EP
    if (idx >= BB * TT * EP) return;
    int e   = idx % EP;
    int row = idx / EP;
    int b = row & 63, t = row >> 6;
    int len = lengths[b];
    int tb  = (t < len) ? (len - 1 - t) : t;
    float vf = 0.f, vb = 0.f;
    if (e < EMB) {
        vf = emb[(long)ids[b * TT + t]  * EMB + e];
        vb = emb[(long)ids[b * TT + tb] * EMB + e];
    }
    Xf[idx] = f2bf(vf);
    Xb[idx] = f2bf(vb);
}

// ---------------------------------------------------------------------------
// Persistent bidirectional LSTM: one workgroup per direction (gridDim.x = 2).
// 512 threads = 16 waves (4 waves/SIMD -> ~256-VGPR budget, no spills).
// Each wave owns 4 tile-sets (mt in 0..3, jt in 0..15); a tile-set keeps the
// 4 gate accumulators (i,f,g,o at n-tiles jt, jt+16, jt+32, jt+48) sharing one
// A fragment, so the i/f/g/o fragments have identical lane->(m,j) mapping and
// the cell state c stays in registers across all 128 steps.
// h lives in LDS (bf16) between steps.
// ---------------------------------------------------------------------------
__global__ __launch_bounds__(512, 1) void k_lstm(
    const unsigned short* __restrict__ Xf, const unsigned short* __restrict__ Xb,
    const unsigned short* __restrict__ WihF, const unsigned short* __restrict__ WihB,
    const unsigned short* __restrict__ WhhF, const unsigned short* __restrict__ WhhB,
    const float* __restrict__ b_f, const float* __restrict__ b_b,
    float* __restrict__ houtF, float* __restrict__ houtB)
{
    const int dir = blockIdx.x;
    const unsigned short* X   = dir ? Xb   : Xf;
    const unsigned short* Wih = dir ? WihB : WihF;
    const unsigned short* Whh = dir ? WhhB : WhhF;
    const float*          bia = dir ? b_b  : b_f;
    float*                hout = dir ? houtB : houtF;

    __shared__ __align__(16) unsigned short hS[BB * RNN];   // 32 KB bf16 h state

    const int tid  = threadIdx.x;
    const int wave = tid >> 5;      // 0..15
    const int lane = tid & 31;
    const int n_lo = lane & 15;
    const int r_off = (lane < 16) ? 0 : 8;

    for (int i = tid; i < BB * RNN; i += 512) hS[i] = 0;    // h0 = 0
    __syncthreads();

    int mt_[4], jt_[4];
    float bb[4][4];
    #pragma unroll
    for (int s = 0; s < 4; ++s) {
        int ts = wave * 4 + s;
        mt_[s] = ts >> 4;
        jt_[s] = ts & 15;
        #pragma unroll
        for (int g = 0; g < 4; ++g) bb[s][g] = bia[g * 256 + jt_[s] * 16 + n_lo];
    }

    float cacc[4][8];
    #pragma unroll
    for (int s = 0; s < 4; ++s)
        #pragma unroll
        for (int r = 0; r < 8; ++r) cacc[s][r] = 0.f;

    #pragma unroll 1
    for (int t = 0; t < TT; ++t) {
        v8f acc[4][4];
        #pragma unroll
        for (int s = 0; s < 4; ++s)
            #pragma unroll
            for (int g = 0; g < 4; ++g) {
                v8f a;
                #pragma unroll
                for (int r = 0; r < 8; ++r) a[r] = 0.f;
                acc[s][g] = a;
            }

        // x_t @ Wih^T  (K = 320); A fragment shared by the 4 gate WMMAs
        #pragma unroll 1
        for (int kt = 0; kt < EP / 32; ++kt) {
            #pragma unroll
            for (int s = 0; s < 4; ++s) {
                v16bf af = load_a_frag(X + (t * BB + mt_[s] * 16) * EP + kt * 32, EP, lane);
                #pragma unroll
                for (int g = 0; g < 4; ++g) {
                    int nt = g * 16 + jt_[s];
                    v16bf bf = load_b_frag(Wih + ((kt * 64) + nt) * 512, lane);
                    acc[s][g] = wmma_bf16(af, bf, acc[s][g]);
                }
            }
        }
        // h_{t-1} @ Whh^T  (K = 256, A from LDS)
        #pragma unroll 1
        for (int kt = 0; kt < RNN / 32; ++kt) {
            #pragma unroll
            for (int s = 0; s < 4; ++s) {
                v16bf af = load_a_frag(hS + (mt_[s] * 16) * RNN + kt * 32, RNN, lane);
                #pragma unroll
                for (int g = 0; g < 4; ++g) {
                    int nt = g * 16 + jt_[s];
                    v16bf bf = load_b_frag(Whh + ((kt * 64) + nt) * 512, lane);
                    acc[s][g] = wmma_bf16(af, bf, acc[s][g]);
                }
            }
        }
        __syncthreads();   // all reads of hS for this step complete

        // gates (PyTorch order i,f,g,o) + state update + write h
        #pragma unroll
        for (int s = 0; s < 4; ++s) {
            int n = jt_[s] * 16 + n_lo;
            #pragma unroll
            for (int r = 0; r < 8; ++r) {
                float gi = acc[s][0][r] + bb[s][0], gf = acc[s][1][r] + bb[s][1];
                float gg = acc[s][2][r] + bb[s][2], go = acc[s][3][r] + bb[s][3];
                float c = sigm(gf) * cacc[s][r] + sigm(gi) * tanhf(gg);
                cacc[s][r] = c;
                float h = sigm(go) * tanhf(c);
                int m = mt_[s] * 16 + r_off + r;
                hS[m * RNN + n] = f2bf(h);
                hout[((long)t * BB + m) * RNN + n] = h;
            }
        }
        __syncthreads();   // hS fully updated before next step's reads
    }
}

// ---------------------------------------------------------------------------
// Build HX [Npad x 512] bf16: concat(h_f[b,t], h_b[b, len-1-t])
// ---------------------------------------------------------------------------
__global__ void k_gather_hx(const int* __restrict__ tok_idx, const int* __restrict__ lengths,
                            const float* __restrict__ hf, const float* __restrict__ hb,
                            unsigned short* __restrict__ HX, int N, int Npad) {
    int idx = blockIdx.x * blockDim.x + threadIdx.x;
    if (idx >= Npad * H2) return;
    int j = idx & (H2 - 1);
    int node = idx >> 9;
    float v = 0.f;
    if (node < N) {
        int tok = tok_idx[node];
        int b = tok >> 7, t = tok & 127;
        if (j < RNN) {
            v = hf[((long)t * BB + b) * RNN + j];
        } else {
            int tbr = lengths[b] - 1 - t;     // un-reverse backward direction
            v = hb[((long)tbr * BB + b) * RNN + (j - RNN)];
        }
    }
    HX[idx] = f2bf(v);
}

// ---------------------------------------------------------------------------
// Tiled WMMA GEMM: C[M x Nout] = A(bf16,row-major,lda) x Bpacked, f32 out.
// One wave per 16x16 tile; block = 4 waves -> 4 consecutive n-tiles.
// ---------------------------------------------------------------------------
__global__ __launch_bounds__(128) void k_gemm(
    const unsigned short* __restrict__ A, int lda,
    const unsigned short* __restrict__ Bp, int NT,
    float* __restrict__ C, int ldc, int KT)
{
    int mt   = blockIdx.x;
    int nt   = blockIdx.y * 4 + (threadIdx.x >> 5);
    int lane = threadIdx.x & 31;
    v8f acc;
    #pragma unroll
    for (int r = 0; r < 8; ++r) acc[r] = 0.f;
    for (int kt = 0; kt < KT; ++kt) {
        v16bf a = load_a_frag(A + (mt * 16) * lda + kt * 32, lda, lane);
        v16bf b = load_b_frag(Bp + ((kt * NT) + nt) * 512, lane);
        acc = wmma_bf16(a, b, acc);
    }
    int n  = nt * 16 + (lane & 15);
    int m0 = mt * 16 + ((lane < 16) ? 0 : 8);
    #pragma unroll
    for (int r = 0; r < 8; ++r) C[(long)(m0 + r) * ldc + n] = acc[r];
}

// ---------------------------------------------------------------------------
// Deterministic CSR construction (children ordered by edge index)
// ---------------------------------------------------------------------------
__global__ void k_zerof(float* p, int n) { int i = blockIdx.x * blockDim.x + threadIdx.x; if (i < n) p[i] = 0.f; }
__global__ void k_zeroi(int*   p, int n) { int i = blockIdx.x * blockDim.x + threadIdx.x; if (i < n) p[i] = 0;   }

__global__ void k_rank(const int* __restrict__ parent, int* __restrict__ rank, int E) {
    int e = blockIdx.x * blockDim.x + threadIdx.x;
    if (e >= E) return;
    int p = parent[e], r = 0;
    for (int q = 0; q < e; ++q) r += (parent[q] == p) ? 1 : 0;
    rank[e] = r;
}
__global__ void k_count(const int* __restrict__ parent, int* __restrict__ counts, int E) {
    int e = blockIdx.x * blockDim.x + threadIdx.x;
    if (e >= E) return;
    atomicAdd(&counts[parent[e]], 1);
}
__global__ void k_scan(const int* __restrict__ counts, int* __restrict__ offsets, int N) {
    if (blockIdx.x | threadIdx.x) return;
    int s = 0;
    for (int i = 0; i < N; ++i) { offsets[i] = s; s += counts[i]; }
    offsets[N] = s;
}
__global__ void k_fill(const int* __restrict__ parent, const int* __restrict__ rank,
                       const int* __restrict__ offsets, int* __restrict__ sorted_e, int E) {
    int e = blockIdx.x * blockDim.x + threadIdx.x;
    if (e >= E) return;
    sorted_e[offsets[parent[e]] + rank[e]] = e;
}

// ---------------------------------------------------------------------------
// Tree-LSTM per-level apply: nodes at node_level == l
// split order is i, o, u (as in reference)
// ---------------------------------------------------------------------------
__global__ void k_apply(int level, const int* __restrict__ node_level,
                        const float* __restrict__ iou, const float* __restrict__ b_iou,
                        const float* __restrict__ c_red,
                        float* __restrict__ c, float* __restrict__ h, int N) {
    int idx = blockIdx.x * blockDim.x + threadIdx.x;
    if (idx >= N * OUT) return;
    int j = idx & (OUT - 1);
    int node = idx >> 8;
    if (node_level[node] != level) return;
    long base = (long)node * G3;
    float i_ = iou[base + j]            + b_iou[j];
    float o_ = iou[base + OUT + j]      + b_iou[OUT + j];
    float u_ = iou[base + 2 * OUT + j]  + b_iou[2 * OUT + j];
    float cn = sigm(i_) * tanhf(u_) + c_red[idx];
    c[idx] = cn;
    h[idx] = sigm(o_) * tanhf(cn);
}

// ---------------------------------------------------------------------------
// Per-level messages: one block (256 threads) per parent. Children with
// node_level == l are reduced sequentially (deterministic). U_iou applied to
// the SUM of child h (linearity); U_f + b_uf per child for the forget gate.
// ---------------------------------------------------------------------------
__global__ __launch_bounds__(256) void k_message(
    int level, const int* __restrict__ offsets, const int* __restrict__ sorted_e,
    const int* __restrict__ child_idx, const int* __restrict__ node_level,
    const float* __restrict__ h, const float* __restrict__ c,
    const float* __restrict__ U_iou, const float* __restrict__ U_f,
    const float* __restrict__ b_uf,
    float* __restrict__ iou, float* __restrict__ c_red)
{
    int p = blockIdx.x;
    int beg = offsets[p], end = offsets[p + 1];
    __shared__ int any;
    __shared__ float hs[OUT];
    __shared__ float hsum[OUT];
    int tid = threadIdx.x;
    if (tid == 0) {
        int a = 0;
        for (int e = beg; e < end; ++e)
            if (node_level[child_idx[sorted_e[e]]] == level) { a = 1; break; }
        any = a;
    }
    __syncthreads();
    if (!any) return;

    float hsum_r = 0.f;
    float credd  = c_red[(long)p * OUT + tid];
    for (int e = beg; e < end; ++e) {
        int ch = child_idx[sorted_e[e]];
        if (node_level[ch] != level) continue;            // uniform per block
        hs[tid] = h[(long)ch * OUT + tid];
        __syncthreads();
        hsum_r += hs[tid];
        float acc = b_uf[tid];
        const float* uf = U_f + (long)tid * OUT;
        for (int k = 0; k < OUT; ++k) acc = fmaf(uf[k], hs[k], acc);
        credd += sigm(acc) * c[(long)ch * OUT + tid];
        __syncthreads();
    }
    c_red[(long)p * OUT + tid] = credd;
    hsum[tid] = hsum_r;
    __syncthreads();
    #pragma unroll
    for (int g = 0; g < 3; ++g) {
        int jj = g * OUT + tid;
        float acc = 0.f;
        const float* ui = U_iou + (long)jj * OUT;
        for (int k = 0; k < OUT; ++k) acc = fmaf(ui[k], hsum[k], acc);
        iou[(long)p * G3 + jj] += acc;
    }
}

// ---------------------------------------------------------------------------
// Host launcher
// ---------------------------------------------------------------------------
extern "C" void kernel_launch(void* const* d_in, const int* in_sizes, int n_in,
                              void* d_out, int out_size, void* d_ws, size_t ws_size,
                              hipStream_t stream) {
    (void)n_in; (void)out_size; (void)ws_size;
    const int*   embed_ids  = (const int*)  d_in[0];
    const int*   lengths    = (const int*)  d_in[1];
    const int*   child_idx  = (const int*)  d_in[2];
    const int*   parent_idx = (const int*)  d_in[3];
    const int*   tok_idx    = (const int*)  d_in[4];
    const int*   node_level = (const int*)  d_in[5];
    /* d_in[6] = n_levels (device scalar, bounded above by 128) */
    const float* emb_table  = (const float*)d_in[7];
    const float* w_ih_f     = (const float*)d_in[8];
    const float* w_hh_f     = (const float*)d_in[9];
    const float* b_f        = (const float*)d_in[10];
    const float* w_ih_b     = (const float*)d_in[11];
    const float* w_hh_b     = (const float*)d_in[12];
    const float* b_b        = (const float*)d_in[13];
    const float* W_iou      = (const float*)d_in[14];
    const float* U_iou      = (const float*)d_in[15];
    const float* b_iou      = (const float*)d_in[16];
    const float* U_f        = (const float*)d_in[17];
    const float* b_uf       = (const float*)d_in[18];
    /* W_hid (19), b_hid (20): dead — h is overwritten by apply at every node */

    const int E = in_sizes[2];          // #edges (~6112)
    const int N = in_sizes[4];          // #nodes (6176, multiple of 16)
    const int Npad = (N + 15) & ~15;
    const int MT = Npad / 16;

    // workspace carve-up
    char* ws = (char*)d_ws;
    size_t off = 0;
    auto carve = [&](size_t bytes) { char* r = ws + off; off = (off + bytes + 255) & ~(size_t)255; return r; };
    unsigned short* Xf    = (unsigned short*)carve((size_t)BB * TT * EP * 2);
    unsigned short* Xb    = (unsigned short*)carve((size_t)BB * TT * EP * 2);
    unsigned short* WihF  = (unsigned short*)carve((size_t)10 * 64 * 512 * 2);
    unsigned short* WihB  = (unsigned short*)carve((size_t)10 * 64 * 512 * 2);
    unsigned short* WhhF  = (unsigned short*)carve((size_t)8  * 64 * 512 * 2);
    unsigned short* WhhB  = (unsigned short*)carve((size_t)8  * 64 * 512 * 2);
    unsigned short* WiouP = (unsigned short*)carve((size_t)16 * 48 * 512 * 2);
    float* hF    = (float*)carve((size_t)BB * TT * RNN * 4);
    float* hB    = (float*)carve((size_t)BB * TT * RNN * 4);
    unsigned short* HX = (unsigned short*)carve((size_t)Npad * H2 * 2);
    float* iou   = (float*)carve((size_t)Npad * G3 * 4);
    float* cbuf  = (float*)carve((size_t)N * OUT * 4);
    float* credb = (float*)carve((size_t)N * OUT * 4);
    int* rankb   = (int*)carve((size_t)E * 4);
    int* counts  = (int*)carve((size_t)N * 4);
    int* offsets = (int*)carve((size_t)(N + 1) * 4);
    int* sortede = (int*)carve((size_t)E * 4);
    float* h_out = (float*)d_out;       // final h [N, 256] lives directly in d_out

    const int TB = 256;
    // 1. embedding gather -> time-major bf16 (both directions)
    k_gather_x<<<(BB * TT * EP + TB - 1) / TB, TB, 0, stream>>>(embed_ids, lengths, emb_table, Xf, Xb);
    // 2. pack weights to WMMA B-fragment layout
    k_pack_b<<<(10 * 64 * 512 + TB - 1) / TB, TB, 0, stream>>>(w_ih_f, WihF, G4, EMB, 10, 64);
    k_pack_b<<<(10 * 64 * 512 + TB - 1) / TB, TB, 0, stream>>>(w_ih_b, WihB, G4, EMB, 10, 64);
    k_pack_b<<<(8  * 64 * 512 + TB - 1) / TB, TB, 0, stream>>>(w_hh_f, WhhF, G4, RNN, 8, 64);
    k_pack_b<<<(8  * 64 * 512 + TB - 1) / TB, TB, 0, stream>>>(w_hh_b, WhhB, G4, RNN, 8, 64);
    k_pack_b<<<(16 * 48 * 512 + TB - 1) / TB, TB, 0, stream>>>(W_iou, WiouP, G3, H2, 16, 48);
    // 3. persistent bidirectional LSTM (WMMA recurrence, one WGP per direction)
    k_lstm<<<2, 512, 0, stream>>>(Xf, Xb, WihF, WihB, WhhF, WhhB, b_f, b_b, hF, hB);
    // 4. node features hx (bf16)
    k_gather_hx<<<(Npad * H2 + TB - 1) / TB, TB, 0, stream>>>(tok_idx, lengths, hF, hB, HX, N, Npad);
    // 5. iou_acc = hx @ W_iou^T (WMMA GEMM, M=Npad, N=768, K=512)
    k_gemm<<<dim3(MT, 48 / 4), 128, 0, stream>>>(HX, H2, WiouP, 48, iou, G3, 16);
    // 6. zero per-call state
    k_zerof<<<(N * OUT + TB - 1) / TB, TB, 0, stream>>>(credb, N * OUT);
    k_zeroi<<<(N + TB - 1) / TB, TB, 0, stream>>>(counts, N);
    // 7. deterministic CSR of children per parent
    k_rank <<<(E + TB - 1) / TB, TB, 0, stream>>>(parent_idx, rankb, E);
    k_count<<<(E + TB - 1) / TB, TB, 0, stream>>>(parent_idx, counts, E);
    k_scan <<<1, 1, 0, stream>>>(counts, offsets, N);
    k_fill <<<(E + TB - 1) / TB, TB, 0, stream>>>(parent_idx, rankb, offsets, sortede, E);
    // 8. topological sweep, leaves -> roots. 128 = provable max #levels for T=128;
    //    levels beyond the true n_levels are empty -> no-ops (deterministic).
    for (int l = 0; l < 128; ++l) {
        k_apply<<<(N * OUT + TB - 1) / TB, TB, 0, stream>>>(l, node_level, iou, b_iou, credb, cbuf, h_out, N);
        k_message<<<N, 256, 0, stream>>>(l, offsets, sortede, child_idx, node_level,
                                         h_out, cbuf, U_iou, U_f, b_uf, iou, credb);
    }
}
